// GatedDeltaNet_13288628814610
// MI455X (gfx1250) — compile-verified, matrix operand order checked by reading, not run on hardware
//
#include <hip/hip_runtime.h>
#include <cstdint>

#define B_SZ   4
#define T_SZ   2048
#define HID_SZ 2048
#define NH_SZ  6
#define DK_SZ  256
#define DV_SZ  512
#define KEYD   1536          // NH*DK
#define VALD   3072          // NH*DV
#define CONVC  6144          // 2*KEYD + VALD
#define PROJ   9228          // CONVC + 2*NH + VALD
#define NPAD1  9344          // ceil(PROJ/128)*128

typedef __attribute__((ext_vector_type(16))) __bf16 v16bf;
typedef __attribute__((ext_vector_type(8)))  float  v8f;

// ---------------------------------------------------------------- helpers ---
__device__ __forceinline__ unsigned short f2bf(float f) {
  unsigned int u = __float_as_uint(f);
  u += 0x7FFFu + ((u >> 16) & 1u);        // round-to-nearest-even
  return (unsigned short)(u >> 16);
}

// Load one 16x32 bf16 WMMA operand fragment from a row-major bf16 matrix.
// CDNA5 layout: lane row = lane&15; lanes 0-15 cover K=[0..7],[16..23],
// lanes 16-31 cover K=[8..15],[24..31]  (relative to k0).
__device__ __forceinline__ v16bf load_frag16(const unsigned short* __restrict__ p,
                                             int ld, int row, int k0, int lane) {
  union { v16bf v; uint4 q[2]; } r;
  const unsigned short* rp = p + (size_t)row * ld + k0 + ((lane & 16) ? 8 : 0);
  r.q[0] = *(const uint4*)(rp);        // K rel +0..7
  r.q[1] = *(const uint4*)(rp + 16);   // K rel +16..23
  return r.v;
}

// ------------------------------------------- f32 -> bf16 (row-padded) -------
// out has padRows rows of ld bf16; rows >= rows are zero-filled.
__global__ __launch_bounds__(256) void cvt_bf16_kernel(const float* __restrict__ in,
                                                       unsigned short* __restrict__ out,
                                                       int rows, int ld, int padRows) {
  const size_t idx8 = ((size_t)blockIdx.x * 256 + threadIdx.x) * 8;
  const size_t total = (size_t)padRows * ld;
  if (idx8 >= total) return;
  const int row = (int)(idx8 / ld);
  union { uint4 q; unsigned short u[8]; } r;
  if (row < rows) {
    const float4 f0 = *(const float4*)(in + idx8);
    const float4 f1 = *(const float4*)(in + idx8 + 4);
    r.u[0] = f2bf(f0.x); r.u[1] = f2bf(f0.y); r.u[2] = f2bf(f0.z); r.u[3] = f2bf(f0.w);
    r.u[4] = f2bf(f1.x); r.u[5] = f2bf(f1.y); r.u[6] = f2bf(f1.z); r.u[7] = f2bf(f1.w);
  } else {
    r.q = make_uint4(0u, 0u, 0u, 0u);
  }
  *(uint4*)(out + idx8) = r.q;
}

// ------------------------------------------------------------ WMMA GEMM -----
// C[m,n] = sum_k A[m,k] * Bm[n,k]   (A: MxK bf16, Bm: NxK bf16, row-major;
// Bm must be padded so every loaded row is valid).  C is f32, ld = N.
// block = 256 threads (8 waves); block tile 128x128; wave tile 32x64.
__global__ __launch_bounds__(256) void gemm_nt_wmma(const unsigned short* __restrict__ A,
                                                    const unsigned short* __restrict__ Bm,
                                                    float* __restrict__ C,
                                                    int M, int N, int K) {
  const int lane  = threadIdx.x & 31;
  const int wid   = threadIdx.x >> 5;
  const int waveM = wid & 3;
  const int waveN = wid >> 2;
  const int mWave = blockIdx.y * 128 + waveM * 32;
  const int nWave = blockIdx.x * 128 + waveN * 64;
  const int lrow  = lane & 15;

  v8f acc[2][4];
#pragma unroll
  for (int i = 0; i < 2; ++i)
#pragma unroll
    for (int j = 0; j < 4; ++j)
#pragma unroll
      for (int e = 0; e < 8; ++e) acc[i][j][e] = 0.f;

  for (int k0 = 0; k0 < K; k0 += 32) {
    v16bf afr[2];
    afr[0] = load_frag16(A, K, mWave + lrow,      k0, lane);
    afr[1] = load_frag16(A, K, mWave + 16 + lrow, k0, lane);
    v16bf bfr[4];
#pragma unroll
    for (int j = 0; j < 4; ++j)
      bfr[j] = load_frag16(Bm, K, nWave + j * 16 + lrow, k0, lane);
#pragma unroll
    for (int i = 0; i < 2; ++i)
#pragma unroll
      for (int j = 0; j < 4; ++j)
        acc[i][j] = __builtin_amdgcn_wmma_f32_16x16x32_bf16(
            false, afr[i], false, bfr[j], (short)0, acc[i][j], false, false);
  }

  const int rOff = (lane & 16) ? 8 : 0;
#pragma unroll
  for (int i = 0; i < 2; ++i) {
#pragma unroll
    for (int j = 0; j < 4; ++j) {
      int col = nWave + j * 16 + lrow;
      if (col < N) {
#pragma unroll
        for (int vv = 0; vv < 8; ++vv) {
          int rowm = mWave + i * 16 + rOff + vv;
          C[(size_t)rowm * N + col] = acc[i][j][vv];
        }
      }
    }
  }
}

// --------------------------------------------- causal conv (K=4) + SiLU -----
__global__ __launch_bounds__(256) void conv_silu_kernel(const float* __restrict__ x,
                                                        const float* __restrict__ w,
                                                        float* __restrict__ qb,
                                                        float* __restrict__ kb,
                                                        float* __restrict__ vb) {
  const int c  = blockIdx.x * 256 + threadIdx.x;   // channel 0..CONVC-1
  const int bt = blockIdx.y;                       // b*T + t
  const int t  = bt & (T_SZ - 1);
  const float4 wc = *(const float4*)(w + (size_t)c * 4);
  float acc;
  if (t >= 3) {
    acc = x[(size_t)(bt - 3) * PROJ + c] * wc.x +
          x[(size_t)(bt - 2) * PROJ + c] * wc.y +
          x[(size_t)(bt - 1) * PROJ + c] * wc.z +
          x[(size_t)(bt    ) * PROJ + c] * wc.w;
  } else {
    const float wj[4] = {wc.x, wc.y, wc.z, wc.w};
    acc = 0.f;
#pragma unroll
    for (int j = 0; j < 4; ++j)
      if (t + j - 3 >= 0) acc += x[(size_t)(bt + j - 3) * PROJ + c] * wj[j];
  }
  float y = acc / (1.f + __expf(-acc));            // SiLU
  if (c < KEYD)          qb[(size_t)bt * KEYD + c] = y;
  else if (c < 2 * KEYD) kb[(size_t)bt * KEYD + (c - KEYD)] = y;
  else                   vb[(size_t)bt * VALD + (c - 2 * KEYD)] = y;
}

// ------------------------------------- l2norm(q,k) + beta/g gate scalars ----
__global__ __launch_bounds__(256) void normalize_gates_kernel(const float* __restrict__ x,
                                                              float* __restrict__ qb,
                                                              float* __restrict__ kb,
                                                              const float* __restrict__ A_log,
                                                              const float* __restrict__ dt_bias,
                                                              float* __restrict__ gb,
                                                              float* __restrict__ bb) {
  __shared__ float red[256];
  const int bt = blockIdx.x;
  const int h  = blockIdx.y;
  const int tx = threadIdx.x;
  const size_t off = (size_t)bt * KEYD + h * DK_SZ + tx;
  float qv = qb[off];
  float kv = kb[off];

  red[tx] = qv * qv;
  __syncthreads();
  for (int s = 128; s > 0; s >>= 1) { if (tx < s) red[tx] += red[tx + s]; __syncthreads(); }
  float qs = rsqrtf(red[0] + 1e-6f);
  __syncthreads();

  red[tx] = kv * kv;
  __syncthreads();
  for (int s = 128; s > 0; s >>= 1) { if (tx < s) red[tx] += red[tx + s]; __syncthreads(); }
  float ks = rsqrtf(red[0] + 1e-6f);

  qb[off] = qv * qs;
  kb[off] = kv * ks;

  if (tx == 0) {
    float av   = x[(size_t)bt * PROJ + CONVC + h];
    float bv   = x[(size_t)bt * PROJ + CONVC + NH_SZ + h];
    float beta = 1.f / (1.f + __expf(-bv));
    float z    = av + dt_bias[h];
    float sp   = (z > 30.f) ? z : log1pf(__expf(z));
    gb[(size_t)bt * NH_SZ + h] = -__expf(A_log[h]) * sp;
    bb[(size_t)bt * NH_SZ + h] = beta;
  }
}

// ------------------------------------------------- gated delta-rule scan ----
// grid = (B*NH, DV/64); 256 threads; state S[256 x 64] in VGPRs (64/thread).
// thread tx: column vcol = tx&63, K-slice ks = tx>>6 (rows ks*64..ks*64+63).
__global__ __launch_bounds__(256) void scan_kernel(const float* __restrict__ q,
                                                   const float* __restrict__ k,
                                                   const float* __restrict__ v,
                                                   const float* __restrict__ g,
                                                   const float* __restrict__ beta,
                                                   float* __restrict__ o) {
  const int bh    = blockIdx.x;
  const int chunk = blockIdx.y;
  const int tx    = threadIdx.x;
  const int vcol  = tx & 63;
  const int ks    = tx >> 6;
  const int b     = bh / NH_SZ;
  const int h     = bh % NH_SZ;
  const float scale = 0.0625f;                   // DK^-0.5

  __shared__ float Kl[256], Ql[256], Vv[64], Pm[256], Po[256], Del[64];

  float S[64];
#pragma unroll
  for (int i = 0; i < 64; ++i) S[i] = 0.f;

  for (int t = 0; t < T_SZ; ++t) {
    const size_t base = ((size_t)(b * T_SZ + t)) * NH_SZ + h;
    Kl[tx] = k[base * DK_SZ + tx];
    Ql[tx] = q[base * DK_SZ + tx];
    if (tx < 64) Vv[tx] = v[base * DV_SZ + chunk * 64 + tx];
    const float eg  = __expf(g[base]);
    const float bt_ = beta[base];
    __syncthreads();

    // mem partials: k . S_old  (decay folded in after reduction)
    float pm = 0.f;
#pragma unroll
    for (int i = 0; i < 64; ++i) pm = __fmaf_rn(Kl[ks * 64 + i], S[i], pm);
    Pm[tx] = pm;
    __syncthreads();

    if (tx < 64) {
      float mem = eg * (Pm[tx] + Pm[64 + tx] + Pm[128 + tx] + Pm[192 + tx]);
      Del[tx] = (Vv[tx] - mem) * bt_;
    }
    __syncthreads();

    // S_new = eg*S_old + k (x) delta ; accumulate o = q . S_new in same pass
    const float d = Del[vcol];
    float po = 0.f;
#pragma unroll
    for (int i = 0; i < 64; ++i) {
      float s = __fmaf_rn(eg, S[i], Kl[ks * 64 + i] * d);
      S[i] = s;
      po = __fmaf_rn(Ql[ks * 64 + i], s, po);
    }
    Po[tx] = po;
    __syncthreads();

    if (tx < 64)
      o[base * DV_SZ + chunk * 64 + tx] =
          scale * (Po[tx] + Po[64 + tx] + Po[128 + tx] + Po[192 + tx]);
  }
}

// ------------- RMS norm + norm_w * SiLU(gate), emit bf16 for GEMM2 ----------
__global__ __launch_bounds__(256) void out_norm_gate_kernel(const float* __restrict__ o,
                                                            const float* __restrict__ x,
                                                            const float* __restrict__ norm_w,
                                                            unsigned short* __restrict__ on16) {
  __shared__ float red[256];
  const int bt = blockIdx.x;
  const int h  = blockIdx.y;
  const int tx = threadIdx.x;
  const size_t base = ((size_t)bt * NH_SZ + h) * DV_SZ;
  float o0 = o[base + tx];
  float o1 = o[base + tx + 256];
  red[tx] = o0 * o0 + o1 * o1;
  __syncthreads();
  for (int s = 128; s > 0; s >>= 1) { if (tx < s) red[tx] += red[tx + s]; __syncthreads(); }
  float r = rsqrtf(red[0] * (1.f / DV_SZ) + 1e-5f);

  const size_t gbase = (size_t)bt * PROJ + CONVC + 2 * NH_SZ + (size_t)h * DV_SZ;
  float g0 = x[gbase + tx];
  float g1 = x[gbase + tx + 256];
  float s0 = g0 / (1.f + __expf(-g0));
  float s1 = g1 / (1.f + __expf(-g1));
  on16[base + tx]       = f2bf(o0 * r * norm_w[tx]       * s0);
  on16[base + tx + 256] = f2bf(o1 * r * norm_w[tx + 256] * s1);
}

// ---------------------------------------------------------------- launch ----
extern "C" void kernel_launch(void* const* d_in, const int* in_sizes, int n_in,
                              void* d_out, int out_size, void* d_ws, size_t ws_size,
                              hipStream_t stream) {
  (void)in_sizes; (void)n_in; (void)out_size; (void)ws_size;
  const float* hs      = (const float*)d_in[0];
  const float* Wqkv    = (const float*)d_in[1];
  const float* convw   = (const float*)d_in[2];
  const float* A_log   = (const float*)d_in[3];
  const float* dt_bias = (const float*)d_in[4];
  const float* norm_w  = (const float*)d_in[5];
  const float* W_o     = (const float*)d_in[6];
  float* out = (float*)d_out;
  float* ws  = (float*)d_ws;

  const size_t BT = (size_t)B_SZ * T_SZ;           // 8192

  // fp32 region
  float* x  = ws;                                  // BT*PROJ
  float* qb = x  + BT * PROJ;                      // BT*KEYD
  float* kb = qb + BT * KEYD;                      // BT*KEYD
  float* vb = kb + BT * KEYD;                      // BT*VALD
  float* ob = vb + BT * VALD;                      // BT*VALD
  float* gb = ob + BT * VALD;                      // BT*NH
  float* bb = gb + BT * NH_SZ;                     // BT*NH
  // bf16 region (all element counts multiples of 8 -> 16B aligned)
  unsigned short* hs16 = (unsigned short*)(bb + BT * NH_SZ); // BT*HID
  unsigned short* w116 = hs16 + BT * HID_SZ;                 // NPAD1*HID (zero-padded)
  unsigned short* on16 = w116 + (size_t)NPAD1 * HID_SZ;      // BT*VALD
  unsigned short* wo16 = on16 + BT * VALD;                   // HID*VALD

  dim3 blk(256);

  // 0) one-shot bf16 conversions (streaming, trivially bandwidth-bound)
  cvt_bf16_kernel<<<(int)((BT * HID_SZ) / 8 / 256), blk, 0, stream>>>(
      hs, hs16, (int)BT, HID_SZ, (int)BT);
  cvt_bf16_kernel<<<(int)(((size_t)NPAD1 * HID_SZ) / 8 / 256), blk, 0, stream>>>(
      Wqkv, w116, PROJ, HID_SZ, NPAD1);
  cvt_bf16_kernel<<<(int)(((size_t)HID_SZ * VALD) / 8 / 256), blk, 0, stream>>>(
      W_o, wo16, HID_SZ, VALD, HID_SZ);

  // 1) x = hs @ Wqkvᵀ  (8192x2048 . 2048x9228), B padded to 9344 rows
  gemm_nt_wmma<<<dim3(NPAD1 / 128, (int)(BT / 128)), blk, 0, stream>>>(
      hs16, w116, x, (int)BT, PROJ, HID_SZ);

  // 2) causal conv + SiLU -> q,k,v raw
  conv_silu_kernel<<<dim3(CONVC / 256, (int)BT), blk, 0, stream>>>(x, convw, qb, kb, vb);

  // 3) l2norm q,k + gate scalars g,beta
  normalize_gates_kernel<<<dim3((int)BT, NH_SZ), blk, 0, stream>>>(
      x, qb, kb, A_log, dt_bias, gb, bb);

  // 4) gated delta scan (state in VGPRs, 192 independent workgroups)
  scan_kernel<<<dim3(B_SZ * NH_SZ, DV_SZ / 64), blk, 0, stream>>>(qb, kb, vb, gb, bb, ob);

  // 5) RMS norm + norm_w * SiLU(gate), emit bf16
  out_norm_gate_kernel<<<dim3((int)BT, NH_SZ), blk, 0, stream>>>(ob, x, norm_w, on16);

  // 6) out = on @ W_oᵀ  (8192x3072 . 3072x2048)
  gemm_nt_wmma<<<dim3(HID_SZ / 128, (int)(BT / 128)), blk, 0, stream>>>(
      on16, wo16, out, (int)BT, HID_SZ, VALD);
}
